// MaskedMultiheadAttention_34995393528439
// MI455X (gfx1250) — matrix operand order repaired and checked by reading, not compiled, not run on hardware
//
#include <hip/hip_runtime.h>

// MI455X / gfx1250: wave32, WMMA 16x16x32 bf16 with f32 accumulate.
// N=4, L=1024, E=1024, H=16, D=64.

typedef __attribute__((ext_vector_type(16))) __bf16 v16bf;
typedef __attribute__((ext_vector_type(8)))  __bf16 v8bf;
typedef __attribute__((ext_vector_type(2)))  __bf16 v2bf;
typedef __attribute__((ext_vector_type(8)))  float  v8f;

static __device__ inline v16bf cat8(v8bf lo, v8bf hi) {
  return __builtin_shufflevector(lo, hi, 0,1,2,3,4,5,6,7,8,9,10,11,12,13,14,15);
}

static __device__ inline v8f wmma_bf16(v16bf a, v16bf b, v8f c) {
  return __builtin_amdgcn_wmma_f32_16x16x32_bf16(false, a, false, b, (short)0, c,
                                                 false, false);
}

// Convert 8 contiguous floats -> one 16B bf16 LDS store (v_cvt_pk + ds_store_b128)
static __device__ inline void st_cvt8(__bf16* dst, const float* s) {
  v8bf t;
  for (int j = 0; j < 8; ++j) t[j] = (__bf16)s[j];
  *(v8bf*)dst = t;
}

// Pack two floats into adjacent bf16 and store as one b32
static __device__ inline void st_pack2(__bf16* dst, float lo, float hi) {
  v2bf t; t[0] = (__bf16)lo; t[1] = (__bf16)hi;
  *(v2bf*)dst = t;
}

// Per-lane async copy 16B global -> LDS (CDNA5 async path, tracked by ASYNCcnt)
static __device__ inline void async_copy16(unsigned lds_off, const __bf16* g) {
  asm volatile("global_load_async_to_lds_b128 %0, %1, off"
               :: "v"(lds_off), "v"(g) : "memory");
}
static __device__ inline void wait_async0() {
  asm volatile("s_wait_asynccnt 0" ::: "memory");
}

// A fragment (16x32 bf16): lane holds row (lane&15); K = half*8+j and 16+half*8+j.
static __device__ inline v16bf load_a_frag(const __bf16* base, int stride) {
  const int lane = threadIdx.x & 31;
  const __bf16* p = base + (lane & 15) * stride + ((lane >> 4) << 3);
  return cat8(*(const v8bf*)(p), *(const v8bf*)(p + 16));
}

// B fragment (32x16): lane holds column (lane&15); K = half*8+j and 16+half*8+j.
// Source stored K-contiguous per column: bT[col][k].
static __device__ inline v16bf load_bT_frag(const __bf16* base, int stride) {
  const int lane = threadIdx.x & 31;
  const __bf16* p = base + (lane & 15) * stride + ((lane >> 4) << 3);
  return cat8(*(const v8bf*)(p), *(const v8bf*)(p + 16));
}

// ---------------------------------------------------------------------------
// Kernel 1: qkv = concat(Q,K,V) @ W + b
// Scatter: q,k -> (N,H,L,D) bf16 ; v -> transposed (N,H,D,L) bf16.
// M = 4096 rows, K = N = 3072. 128x128 block tile, 8 waves of 32x64.
// ---------------------------------------------------------------------------
__global__ __launch_bounds__(256)
void qkv_proj_kernel(const float* __restrict__ Qin, const float* __restrict__ Kin,
                     const float* __restrict__ Vin, const float* __restrict__ W,
                     const float* __restrict__ bias,
                     __bf16* __restrict__ q_ws, __bf16* __restrict__ k_ws,
                     __bf16* __restrict__ v_ws) {
  __shared__ __bf16 As[128][40];     // padded: rows 80B apart -> 16B aligned chunks
  __shared__ __bf16 BsT[128][40];    // BsT[col][k]
  const int tid  = threadIdx.x;
  const int lane = tid & 31;
  const int wid  = tid >> 5;
  const int wr   = wid >> 1;         // 0..3 : 32-row slab
  const int wc   = wid & 1;          // 0..1 : 64-col slab
  const int m0   = blockIdx.y * 128;
  const int n0   = blockIdx.x * 128;

  const v8f zacc = {0.f,0.f,0.f,0.f,0.f,0.f,0.f,0.f};
  v8f acc[2][4];
  for (int mi = 0; mi < 2; ++mi)
    for (int ni = 0; ni < 4; ++ni) acc[mi][ni] = zacc;

  const int arow = tid >> 1;               // 0..127
  const int acb  = (tid & 1) << 4;         // 0 or 16
  const int bn   = (m0 + arow) >> 10;      // batch
  const int bl   = (m0 + arow) & 1023;     // seq position
  const int bkp  = (tid >> 4) << 1;        // 0,2,..,30 : W row pair within K-tile
  const int bcb  = (tid & 15) << 3;        // 0..120    : W col base (8 cols)

  for (int k0 = 0; k0 < 3072; k0 += 32) {
    {   // A tile: X = concat(queries, keys, values) along feature dim
      const int c = k0 + acb;
      const int region = c >> 10;          // 0:q 1:k 2:v (chunks never straddle)
      const float* src = (region == 0) ? Qin : (region == 1) ? Kin : Vin;
      const float* p = src + (bn * 1024 + bl) * 1024 + (c & 1023);
      st_cvt8(&As[arow][acb], p);
      st_cvt8(&As[arow][acb + 8], p + 8);
    }
    {   // B tile: W[k][c] -> BsT[c][k]; packed pair stores along k
      const float* p0 = W + (k0 + bkp) * 3072 + n0 + bcb;
      const float* p1 = p0 + 3072;
      for (int j = 0; j < 8; ++j)
        st_pack2(&BsT[bcb + j][bkp], p0[j], p1[j]);
    }
    if (k0 + 64 < 3072)                     // global_prefetch_b8 of W two K-tiles ahead
      __builtin_prefetch(W + (size_t)(k0 + 64) * 3072 + n0 + (tid << 4), 0, 1);
    __syncthreads();

    v16bf af[2], bfr[4];
    for (int mi = 0; mi < 2; ++mi) af[mi]  = load_a_frag(&As[wr*32 + mi*16][0], 40);
    for (int ni = 0; ni < 4; ++ni) bfr[ni] = load_bT_frag(&BsT[wc*64 + ni*16][0], 40);
    for (int mi = 0; mi < 2; ++mi)
      for (int ni = 0; ni < 4; ++ni)
        acc[mi][ni] = wmma_bf16(af[mi], bfr[ni], acc[mi][ni]);
    __syncthreads();
  }

  // Epilogue: +bias, scatter (q,k head-major; v transposed head-major)
  const int halfsel = (lane >> 4) << 3;
  for (int mi = 0; mi < 2; ++mi)
    for (int ni = 0; ni < 4; ++ni)
      for (int r = 0; r < 8; ++r) {
        const int row  = m0 + wr*32 + mi*16 + r + halfsel;
        const int col  = n0 + wc*64 + ni*16 + (lane & 15);
        const int n    = row >> 10;
        const int lpos = row & 1023;
        const float val = acc[mi][ni][r] + bias[col];
        const int region = col >> 10;
        const int cr = col & 1023;
        const int h = cr >> 6, d = cr & 63;
        if (region == 2) {                 // V stored (N,H,D,L) for contiguous PV
          v_ws[((n*16 + h)*64 + d)*1024 + lpos] = (__bf16)val;
        } else {
          __bf16* dst = (region == 0) ? q_ws : k_ws;
          dst[((n*16 + h)*1024 + lpos)*64 + d] = (__bf16)val;
        }
      }
}

// ---------------------------------------------------------------------------
// Kernel 2: flash attention. One wave per (n, h, 16-query tile).
// S = Q K^T (2 WMMAs per 16-key tile, D=64), masks -> -1e3, scale 1/32,
// online softmax, P V via LDS round-trip to A-fragment layout.
// ---------------------------------------------------------------------------
__global__ __launch_bounds__(32)
void flash_attn_kernel(const __bf16* __restrict__ q_ws, const __bf16* __restrict__ k_ws,
                       const __bf16* __restrict__ v_ws, const int* __restrict__ amask,
                       const int* __restrict__ pmask, __bf16* __restrict__ ctx) {
  __shared__ __bf16 P[16][40];
  const int b     = blockIdx.x;
  const int qt    = b & 63;
  const int h     = (b >> 6) & 15;
  const int n     = b >> 10;
  const int lane  = threadIdx.x;
  const int half  = lane >> 4;
  const int col16 = lane & 15;

  // Q fragments (held for the whole key loop)
  const __bf16* qb = q_ws + ((n*16 + h)*1024 + qt*16 + col16) * 64;
  const v16bf aq0 = cat8(*(const v8bf*)(qb +      half*8), *(const v8bf*)(qb + 16 + half*8));
  const v16bf aq1 = cat8(*(const v8bf*)(qb + 32 + half*8), *(const v8bf*)(qb + 48 + half*8));

  const v8f z = {0.f,0.f,0.f,0.f,0.f,0.f,0.f,0.f};
  v8f o0 = z, o1 = z, o2 = z, o3 = z;
  float mrun[8], lrun[8];
  for (int r = 0; r < 8; ++r) { mrun[r] = -1e30f; lrun[r] = 0.f; }

  const __bf16* kh = k_ws + (n*16 + h)*1024*64;     // (L, D)
  const __bf16* vT = v_ws + (n*16 + h)*64*1024;     // (D, L) transposed

  for (int kb = 0; kb < 1024; kb += 32) {
    // K fragments: lane = key column, D = half*8+j and 16+half*8+j
    const __bf16* kp0 = kh + (kb + col16) * 64 + half*8;
    const __bf16* kp1 = kp0 + 16*64;
    const v16bf bk00 = cat8(*(const v8bf*)(kp0),      *(const v8bf*)(kp0 + 16));
    const v16bf bk01 = cat8(*(const v8bf*)(kp0 + 32), *(const v8bf*)(kp0 + 48));
    const v16bf bk10 = cat8(*(const v8bf*)(kp1),      *(const v8bf*)(kp1 + 16));
    const v16bf bk11 = cat8(*(const v8bf*)(kp1 + 32), *(const v8bf*)(kp1 + 48));

    v8f s0 = z, s1 = z;
    s0 = wmma_bf16(aq0, bk00, s0);
    s0 = wmma_bf16(aq1, bk01, s0);
    s1 = wmma_bf16(aq0, bk10, s1);
    s1 = wmma_bf16(aq1, bk11, s1);

    // masks + scale + online softmax (per C/D row r+8*half)
    const int c0 = kb + col16, c1 = c0 + 16;
    const int p0 = pmask[n*1024 + c0];
    const int p1 = pmask[n*1024 + c1];
    float alpha[8];
    for (int r = 0; r < 8; ++r) {
      const int qrow = qt*16 + r + half*8;
      float e0 = (amask[qrow*1024 + c0] == 0 || p0 == 0) ? -1000.f : s0[r];
      float e1 = (amask[qrow*1024 + c1] == 0 || p1 == 0) ? -1000.f : s1[r];
      e0 *= 0.03125f; e1 *= 0.03125f;            // 1/sqrt(E), applied post-mask
      float t = fmaxf(e0, e1);
      t = fmaxf(t, __shfl_xor(t, 1, 32));
      t = fmaxf(t, __shfl_xor(t, 2, 32));
      t = fmaxf(t, __shfl_xor(t, 4, 32));
      t = fmaxf(t, __shfl_xor(t, 8, 32));        // stays within 16-lane half
      const float mnew = fmaxf(mrun[r], t);
      alpha[r] = __expf(mrun[r] - mnew);
      mrun[r] = mnew;
      e0 = __expf(e0 - mnew);
      e1 = __expf(e1 - mnew);
      float rs = e0 + e1;
      rs += __shfl_xor(rs, 1, 32);
      rs += __shfl_xor(rs, 2, 32);
      rs += __shfl_xor(rs, 4, 32);
      rs += __shfl_xor(rs, 8, 32);
      lrun[r] = lrun[r] * alpha[r] + rs;
      s0[r] = e0; s1[r] = e1;
    }
    for (int r = 0; r < 8; ++r) {
      o0[r] *= alpha[r]; o1[r] *= alpha[r]; o2[r] *= alpha[r]; o3[r] *= alpha[r];
    }

    // P (C/D layout) -> LDS -> A-fragment layout
    __syncthreads();
    for (int r = 0; r < 8; ++r) {
      P[r + half*8][col16]      = (__bf16)s0[r];
      P[r + half*8][16 + col16] = (__bf16)s1[r];
    }
    __syncthreads();
    const v16bf ap = load_a_frag(&P[0][0], 40);

    // O += P * V  (4 output column tiles; V transposed -> contiguous B loads)
    for (int t4 = 0; t4 < 4; ++t4) {
      const __bf16* vp = vT + (t4*16 + col16)*1024 + kb + half*8;
      const v16bf bv = cat8(*(const v8bf*)(vp), *(const v8bf*)(vp + 16));
      v8f* op = (t4 == 0) ? &o0 : (t4 == 1) ? &o1 : (t4 == 2) ? &o2 : &o3;
      *op = wmma_bf16(ap, bv, *op);
    }
  }

  // normalize and write context (N, L, E) with E = h*64 + d
  for (int t4 = 0; t4 < 4; ++t4) {
    const v8f ov = (t4 == 0) ? o0 : (t4 == 1) ? o1 : (t4 == 2) ? o2 : o3;
    for (int r = 0; r < 8; ++r) {
      const int row = qt*16 + r + half*8;
      const int col = h*64 + t4*16 + col16;
      ctx[(n*1024 + row)*1024 + col] = (__bf16)(ov[r] / lrun[r]);
    }
  }
}

// ---------------------------------------------------------------------------
// Kernel 3: out = ctx @ fc_w^T + fc_b   (4096x1024 x 1024x1024, fp32 out)
// ctx is bf16 already -> A tile staged with global_load_async_to_lds_b128.
// B[k][c] = fc_w[c][k]  ->  BsT[c][k] is a contiguous row read of fc_w.
// ---------------------------------------------------------------------------
__global__ __launch_bounds__(256)
void fc_out_kernel(const __bf16* __restrict__ ctx, const float* __restrict__ Wf,
                   const float* __restrict__ bf_, float* __restrict__ out) {
  __shared__ __bf16 As[128][40];
  __shared__ __bf16 BsT[128][40];
  const int tid  = threadIdx.x;
  const int lane = tid & 31;
  const int wid  = tid >> 5;
  const int wr   = wid >> 1;
  const int wc   = wid & 1;
  const int m0   = blockIdx.y * 128;
  const int n0   = blockIdx.x * 128;

  const v8f zacc = {0.f,0.f,0.f,0.f,0.f,0.f,0.f,0.f};
  v8f acc[2][4];
  for (int mi = 0; mi < 2; ++mi)
    for (int ni = 0; ni < 4; ++ni) acc[mi][ni] = zacc;

  const int arow = tid >> 1;
  const int acb  = (tid & 1) << 4;
  const unsigned lds_a = (unsigned)(unsigned long long)&As[arow][acb];

  for (int k0 = 0; k0 < 1024; k0 += 32) {
    // A tile: bf16 ctx -> LDS via per-lane async DMA (no conversion needed)
    const __bf16* ap = ctx + (m0 + arow) * 1024 + k0 + acb;
    async_copy16(lds_a, ap);
    async_copy16(lds_a + 16, ap + 8);
    // B tile: fc_w row read (contiguous), packed bf16 ds_store_b128
    const float* wp = Wf + (n0 + arow) * 1024 + k0 + acb;
    st_cvt8(&BsT[arow][acb], wp);
    st_cvt8(&BsT[arow][acb + 8], wp + 8);
    wait_async0();
    __syncthreads();

    v16bf af[2], bfr[4];
    for (int mi = 0; mi < 2; ++mi) af[mi]  = load_a_frag(&As[wr*32 + mi*16][0], 40);
    for (int ni = 0; ni < 4; ++ni) bfr[ni] = load_bT_frag(&BsT[wc*64 + ni*16][0], 40);
    for (int mi = 0; mi < 2; ++mi)
      for (int ni = 0; ni < 4; ++ni)
        acc[mi][ni] = wmma_bf16(af[mi], bfr[ni], acc[mi][ni]);
    __syncthreads();
  }

  const int halfsel = (lane >> 4) << 3;
  for (int mi = 0; mi < 2; ++mi)
    for (int ni = 0; ni < 4; ++ni)
      for (int r = 0; r < 8; ++r) {
        const int row = m0 + wr*32 + mi*16 + r + halfsel;
        const int col = n0 + wc*64 + ni*16 + (lane & 15);
        out[row * 1024 + col] = acc[mi][ni][r] + bf_[col];
      }
}

// ---------------------------------------------------------------------------
extern "C" void kernel_launch(void* const* d_in, const int* in_sizes, int n_in,
                              void* d_out, int out_size, void* d_ws, size_t ws_size,
                              hipStream_t stream) {
  const float* values  = (const float*)d_in[0];
  const float* keys    = (const float*)d_in[1];
  const float* queries = (const float*)d_in[2];
  const int*   amask   = (const int*)d_in[3];
  const int*   pmask   = (const int*)d_in[4];
  const float* inW     = (const float*)d_in[5];
  const float* inB     = (const float*)d_in[6];
  const float* fcW     = (const float*)d_in[7];
  const float* fcB     = (const float*)d_in[8];
  float* out = (float*)d_out;

  char* ws = (char*)d_ws;
  const size_t REG = 8ull << 20;                 // 8 MiB per bf16 region
  __bf16* q_ws = reinterpret_cast<__bf16*>(ws);
  __bf16* k_ws = reinterpret_cast<__bf16*>(ws + REG);
  __bf16* v_ws = reinterpret_cast<__bf16*>(ws + 2*REG);   // transposed (N,H,D,L)
  __bf16* ctx  = reinterpret_cast<__bf16*>(ws + 3*REG);

  // 1) fused input projection (concat order: queries, keys, values)
  qkv_proj_kernel<<<dim3(24, 32), 256, 0, stream>>>(queries, keys, values, inW, inB,
                                                    q_ws, k_ws, v_ws);
  // 2) masked flash attention: one wave per (n, h, 16-query tile)
  flash_attn_kernel<<<dim3(4 * 16 * 64), 32, 0, stream>>>(q_ws, k_ws, v_ws,
                                                          amask, pmask, ctx);
  // 3) output projection
  fc_out_kernel<<<dim3(8, 32), 256, 0, stream>>>(ctx, fcW, fcB, out);
}
